// GINMoleculeNet_8237747274041
// MI455X (gfx1250) — matrix operand order repaired and checked by reading, not compiled, not run on hardware
//
#include <hip/hip_runtime.h>

typedef __attribute__((ext_vector_type(16))) _Float16 v16h;
typedef __attribute__((ext_vector_type(4)))  _Float16 h4;
typedef __attribute__((ext_vector_type(8)))  float    v8f;
typedef __attribute__((ext_vector_type(4)))  float    v4f;

// Problem constants (fixed by the reference)
#define GN   100000
#define GE   300000
#define FIN  128
#define GH   256
#define GG   4000
#define GL   5

// GEMM tiling
#define TM   128
#define TN   128
#define TK   32
#define LST  40     // LDS row stride in f16 (32 data + 8 pad)

// ---------------------------------------------------------------------------
// One-time: Wt[n][k] (f16) = W[k][n] (f32). Weight matrices are tiny.
// ---------------------------------------------------------------------------
__global__ void transpose_w_f16(const float* __restrict__ W, _Float16* __restrict__ Wt,
                                int K, int Ncols)
{
    const int idx = blockIdx.x * 256 + threadIdx.x;
    if (idx < K * Ncols) {
        const int k = idx / Ncols;
        const int n = idx % Ncols;
        Wt[(size_t)n * K + k] = (_Float16)W[idx];
    }
}

// ---------------------------------------------------------------------------
// Fused GEMM: Y = [relu?]( (affine?(relu(scale*A+shift)):A) @ W + bias )
// A: [M,K] f32 row-major.  Wt: [Ncols,K] f16 (pre-transposed weights).
// Optionally accumulates per-column sum / sum-of-squares for BatchNorm stats.
// Register-prefetch pipeline: next tile's global loads issued before the MMAs.
// ---------------------------------------------------------------------------
__global__ __launch_bounds__(256)
void gemm_bn_wmma(const float* __restrict__ A, const _Float16* __restrict__ Wt,
                  const float* __restrict__ bias,
                  const float* __restrict__ in_scale, const float* __restrict__ in_shift,
                  float* __restrict__ Y,
                  float* __restrict__ colsum, float* __restrict__ colsq,
                  int M, int K, int Ncols, int relu_out)
{
    __shared__ __align__(16) _Float16 aT[TM * LST];
    __shared__ __align__(16) _Float16 bT[TN * LST];

    const int tid  = threadIdx.x;
    const int lane = tid & 31;
    const int wave = tid >> 5;
    const int rowBase = blockIdx.x * TM;
    const int colBase = blockIdx.y * TN;
    const int waveRow = (wave >> 1) * 32;   // 4 wave-rows of 32
    const int waveCol = (wave & 1) * 64;    // 2 wave-cols of 64

    // staging coordinates
    const int akq = (tid & 7) * 4;          // A: 4 consecutive k per thread
    const int ar0 = tid >> 3;               // A: base row (0..31), 4 rows stride 32
    const int bn  = tid >> 1;               // B: row (= output col) 0..127
    const int bkh = (tid & 1) * 16;         // B: k half 0/16
    const _Float16* wrow = Wt + (size_t)(colBase + bn) * K + bkh;

    v8f acc[2][4];
    #pragma unroll
    for (int i = 0; i < 2; ++i)
        #pragma unroll
        for (int j = 0; j < 4; ++j)
            #pragma unroll
            for (int e = 0; e < 8; ++e) acc[i][j][e] = 0.0f;

    float4 aReg[4];
    v4f    bReg0, bReg1;

    auto loadTiles = [&](int k0) {
        #pragma unroll
        for (int rr = 0; rr < 4; ++rr) {
            const int grow = rowBase + ar0 + rr * 32;
            float4 v = make_float4(0.f, 0.f, 0.f, 0.f);
            if (grow < M) {
                v = *(const float4*)(A + (size_t)grow * K + k0 + akq);
                if (in_scale) {
                    const int kk = k0 + akq;
                    v.x = fmaxf(fmaf(in_scale[kk + 0], v.x, in_shift[kk + 0]), 0.f);
                    v.y = fmaxf(fmaf(in_scale[kk + 1], v.y, in_shift[kk + 1]), 0.f);
                    v.z = fmaxf(fmaf(in_scale[kk + 2], v.z, in_shift[kk + 2]), 0.f);
                    v.w = fmaxf(fmaf(in_scale[kk + 3], v.w, in_shift[kk + 3]), 0.f);
                }
            }
            aReg[rr] = v;
        }
        bReg0 = *(const v4f*)(wrow + k0);       // 8 f16
        bReg1 = *(const v4f*)(wrow + k0 + 8);   // 8 f16
    };

    auto storeTiles = [&]() {
        #pragma unroll
        for (int rr = 0; rr < 4; ++rr) {
            h4 hv = { (_Float16)aReg[rr].x, (_Float16)aReg[rr].y,
                      (_Float16)aReg[rr].z, (_Float16)aReg[rr].w };
            *(h4*)&aT[(ar0 + rr * 32) * LST + akq] = hv;       // ds_store_b64
        }
        *(v4f*)&bT[bn * LST + bkh]     = bReg0;                // ds_store_b128
        *(v4f*)&bT[bn * LST + bkh + 8] = bReg1;                // ds_store_b128
    };

    const int nsteps = K / TK;
    loadTiles(0);

    for (int s = 0; s < nsteps; ++s) {
        storeTiles();
        __syncthreads();

        // prefetch next tile's globals; loads retire while WMMAs run
        if (s + 1 < nsteps) loadTiles((s + 1) * TK);

        // ---- per-wave fragment gathers per CDNA5 WMMA VGPR layout, then MMA
        v16h afrag[2];
        {
            const int kb = (lane >> 4) * 8;   // lanes 0-15: K 0-7/16-23; 16-31: K 8-15/24-31
            const int rl = lane & 15;
            #pragma unroll
            for (int rt = 0; rt < 2; ++rt) {
                const _Float16* p = &aT[(waveRow + rt * 16 + rl) * LST + kb];
                v4f* d = (v4f*)&afrag[rt];
                d[0] = *(const v4f*)p;
                d[1] = *(const v4f*)(p + 16);
            }
        }
        #pragma unroll
        for (int ct = 0; ct < 4; ++ct) {
            const int kb2 = (lane >> 4) * 8;  // lanes 0-15: K 0-7/16-23; 16-31: K 8-15/24-31
            const _Float16* p = &bT[(waveCol + ct * 16 + (lane & 15)) * LST + kb2];
            v16h bfrag;
            v4f* d = (v4f*)&bfrag;
            d[0] = *(const v4f*)p;
            d[1] = *(const v4f*)(p + 16);
            #pragma unroll
            for (int rt = 0; rt < 2; ++rt) {
                acc[rt][ct] = __builtin_amdgcn_wmma_f32_16x16x32_f16(
                    false, afrag[rt], false, bfrag, (short)0, acc[rt][ct], false, false);
            }
        }
        __syncthreads();
    }

    // ---- epilogue: bias, optional ReLU, store, BN-stat atomics
    #pragma unroll
    for (int rt = 0; rt < 2; ++rt) {
        const int rB = rowBase + waveRow + rt * 16 + ((lane >> 4) << 3);
        #pragma unroll
        for (int ct = 0; ct < 4; ++ct) {
            const int n  = colBase + waveCol + ct * 16 + (lane & 15);
            const float bv = bias[n];
            float s = 0.f, q = 0.f;
            #pragma unroll
            for (int r = 0; r < 8; ++r) {
                const int row = rB + r;
                if (row < M) {
                    float c = acc[rt][ct][r] + bv;
                    if (relu_out) c = fmaxf(c, 0.f);
                    Y[(size_t)row * Ncols + n] = c;
                    s += c; q += c * c;
                }
            }
            if (colsum) {
                atomicAdd(colsum + n, s);
                atomicAdd(colsq + n, q);
            }
        }
    }
}

// ---------------------------------------------------------------------------
// z = (1 + eps[l]) * h       (seed for scatter accumulation)
// ---------------------------------------------------------------------------
__global__ void scale_copy(const float* __restrict__ h, float* __restrict__ z,
                           const float* __restrict__ eps, int l, size_t n)
{
    size_t i = (size_t)blockIdx.x * blockDim.x + threadIdx.x;
    if (i < n) z[i] = (1.0f + eps[l]) * h[i];
}

// z[dst] += h[src] per edge; one block per edge, one lane per feature
__global__ void scatter_add(const float* __restrict__ h, const long long* __restrict__ ei,
                            float* __restrict__ z, int E)
{
    const int e = blockIdx.x;
    const int f = threadIdx.x;
    const int s = (int)ei[e];
    const int d = (int)ei[(size_t)E + e];
    atomicAdd(&z[(size_t)d * GH + f], h[(size_t)s * GH + f]);
}

__global__ void zero_f32(float* __restrict__ p, size_t n)
{
    size_t i = (size_t)blockIdx.x * blockDim.x + threadIdx.x;
    if (i < n) p[i] = 0.0f;
}

// scale = g * rsqrt(var + eps); shift = beta - mean * scale
__global__ void bn_finalize(const float* __restrict__ cs, const float* __restrict__ cq,
                            const float* __restrict__ g, const float* __restrict__ be,
                            float invM, float* __restrict__ scale, float* __restrict__ shift)
{
    const int i = threadIdx.x;
    const float m  = cs[i] * invM;
    const float v  = cq[i] * invM - m * m;
    const float sc = g[i] * rsqrtf(v + 1e-5f);
    scale[i] = sc;
    shift[i] = be[i] - m * sc;
}

// h = relu(scale[f]*y + shift[f])
__global__ void apply_bn_relu(const float* __restrict__ y, const float* __restrict__ scale,
                              const float* __restrict__ shift, float* __restrict__ h, size_t n)
{
    size_t i = (size_t)blockIdx.x * blockDim.x + threadIdx.x;
    if (i < n) {
        const int f = (int)(i & (size_t)(GH - 1));
        h[i] = fmaxf(fmaf(scale[f], y[i], shift[f]), 0.0f);
    }
}

// pooled[batch[n]] += h[n]; cnt[batch[n]] += 1
__global__ void pool_add(const float* __restrict__ h, const long long* __restrict__ batch,
                         float* __restrict__ pooled, float* __restrict__ cnt)
{
    const int n = blockIdx.x;
    const int f = threadIdx.x;
    const int b = (int)batch[n];
    atomicAdd(&pooled[(size_t)b * GH + f], h[(size_t)n * GH + f]);
    if (f == 0) atomicAdd(&cnt[b], 1.0f);
}

__global__ void mean_div(float* __restrict__ pooled, const float* __restrict__ cnt)
{
    const int g = blockIdx.x;
    const int f = threadIdx.x;
    pooled[(size_t)g * GH + f] *= 1.0f / fmaxf(cnt[g], 1.0f);
}

// out[g] = sum_j hid[g][j]*W2[j] + b2
__global__ void head_out(const float* __restrict__ hid, const float* __restrict__ W2,
                         const float* __restrict__ b2, float* __restrict__ out)
{
    __shared__ float red[128];
    const int g = blockIdx.x;
    const int t = threadIdx.x;
    red[t] = hid[(size_t)g * 128 + t] * W2[t];
    __syncthreads();
    #pragma unroll
    for (int s = 64; s > 0; s >>= 1) {
        if (t < s) red[t] += red[t + s];
        __syncthreads();
    }
    if (t == 0) out[g] = red[0] + b2[0];
}

// ---------------------------------------------------------------------------
extern "C" void kernel_launch(void* const* d_in, const int* in_sizes, int n_in,
                              void* d_out, int out_size, void* d_ws, size_t ws_size,
                              hipStream_t stream)
{
    const float*     x     = (const float*)d_in[0];
    const long long* ei    = (const long long*)d_in[1];
    const long long* batch = (const long long*)d_in[2];
    const float*     in_W  = (const float*)d_in[3];
    const float*     in_b  = (const float*)d_in[4];
    const float*     cW1   = (const float*)d_in[5];
    const float*     cb1   = (const float*)d_in[6];
    const float*     cg1   = (const float*)d_in[7];
    const float*     cbe1  = (const float*)d_in[8];
    const float*     cW2   = (const float*)d_in[9];
    const float*     cb2   = (const float*)d_in[10];
    const float*     cg2   = (const float*)d_in[11];
    const float*     cbe2  = (const float*)d_in[12];
    const float*     eps   = (const float*)d_in[13];
    const float*     hW1   = (const float*)d_in[14];
    const float*     hb1   = (const float*)d_in[15];
    const float*     hW2   = (const float*)d_in[16];
    const float*     hb2   = (const float*)d_in[17];
    float*           out   = (float*)d_out;

    const size_t NH = (size_t)GN * GH;
    float* ws     = (float*)d_ws;
    float* h      = ws;                         // [N,H]
    float* z      = ws + NH;                    // [N,H]
    float* y      = ws + 2 * NH;                // [N,H]
    float* pooled = ws + 3 * NH;                // [G,H]
    float* cnt    = pooled + (size_t)GG * GH;   // [G]
    float* cs1    = cnt + GG;                   // 4 contiguous stat arrays of GH
    float* cq1    = cs1 + GH;
    float* cs2    = cq1 + GH;
    float* cq2    = cs2 + GH;
    float* scA    = cq2 + GH;
    float* shA    = scA + GH;
    float* scB    = shA + GH;
    float* shB    = scB + GH;
    float* hid    = shB + GH;                   // [G, H/2]
    // f16 pre-transposed weight arena (16B-aligned by construction)
    _Float16* wtA   = (_Float16*)(hid + (size_t)GG * (GH / 2));
    _Float16* in_Wt = wtA;                                  // [GH,  FIN]
    _Float16* w1t   = in_Wt + (size_t)GH * FIN;             // L x [GH, GH]
    _Float16* w2t   = w1t + (size_t)GL * GH * GH;           // L x [GH, GH]
    _Float16* hW1t  = w2t + (size_t)GL * GH * GH;           // [GH/2, GH]

    const dim3 blk(256);
    const dim3 gridN((GN + TM - 1) / TM, GH / TN);
    const float invN = 1.0f / (float)GN;

    // ---- one-time weight transpose + f16 conversion (tiny)
    transpose_w_f16<<<(FIN * GH + 255) / 256, blk, 0, stream>>>(in_W, in_Wt, FIN, GH);
    for (int l = 0; l < GL; ++l) {
        transpose_w_f16<<<(GH * GH + 255) / 256, blk, 0, stream>>>(
            cW1 + (size_t)l * GH * GH, w1t + (size_t)l * GH * GH, GH, GH);
        transpose_w_f16<<<(GH * GH + 255) / 256, blk, 0, stream>>>(
            cW2 + (size_t)l * GH * GH, w2t + (size_t)l * GH * GH, GH, GH);
    }
    transpose_w_f16<<<(GH * (GH / 2) + 255) / 256, blk, 0, stream>>>(hW1, hW1t, GH, GH / 2);

    // h = relu(x @ in_W + in_b)
    gemm_bn_wmma<<<gridN, blk, 0, stream>>>(x, in_Wt, in_b, nullptr, nullptr,
                                            h, nullptr, nullptr, GN, FIN, GH, 1);

    for (int l = 0; l < GL; ++l) {
        // z = (1+eps)*h + segment_sum(h[src] -> dst)
        scale_copy<<<(unsigned)((NH + 255) / 256), blk, 0, stream>>>(h, z, eps, l, NH);
        scatter_add<<<GE, GH, 0, stream>>>(h, ei, z, GE);

        // zero BN stat accumulators (cs1,cq1,cs2,cq2 contiguous)
        zero_f32<<<(4 * GH + 255) / 256, blk, 0, stream>>>(cs1, (size_t)(4 * GH));

        // y = z @ W1 + b1 (collect col stats)
        gemm_bn_wmma<<<gridN, blk, 0, stream>>>(z, w1t + (size_t)l * GH * GH, cb1 + l * GH,
                                                nullptr, nullptr, y, cs1, cq1, GN, GH, GH, 0);
        bn_finalize<<<1, GH, 0, stream>>>(cs1, cq1, cg1 + l * GH, cbe1 + l * GH, invN, scA, shA);

        // z = relu(bn1(y)) @ W2 + b2 (bn1+relu fused into A load; collect col stats)
        gemm_bn_wmma<<<gridN, blk, 0, stream>>>(y, w2t + (size_t)l * GH * GH, cb2 + l * GH,
                                                scA, shA, z, cs2, cq2, GN, GH, GH, 0);
        bn_finalize<<<1, GH, 0, stream>>>(cs2, cq2, cg2 + l * GH, cbe2 + l * GH, invN, scB, shB);

        // h = relu(bn2(z))
        apply_bn_relu<<<(unsigned)((NH + 255) / 256), blk, 0, stream>>>(z, scB, shB, h, NH);
    }

    // global mean pool
    zero_f32<<<(unsigned)(((size_t)GG * GH + GG + 255) / 256), blk, 0, stream>>>(
        pooled, (size_t)GG * GH + GG);
    pool_add<<<GN, GH, 0, stream>>>(h, batch, pooled, cnt);
    mean_div<<<GG, GH, 0, stream>>>(pooled, cnt);

    // head: hid = relu(pooled @ hW1 + hb1)  [4000,256]x[256,128]
    {
        const dim3 gridH((GG + TM - 1) / TM, (GH / 2) / TN);
        gemm_bn_wmma<<<gridH, blk, 0, stream>>>(pooled, hW1t, hb1, nullptr, nullptr,
                                                hid, nullptr, nullptr, GG, GH, GH / 2, 1);
    }
    // out = hid @ hW2 + hb2
    head_out<<<GG, GH / 2, 0, stream>>>(hid, hW2, hb2, out);
}
